// QuantizedLinear_23373212025217
// MI455X (gfx1250) — compile-verified, matrix operand order checked by reading, not run on hardware
//
#include <hip/hip_runtime.h>
#include <stdint.h>

typedef _Float16 half8 __attribute__((ext_vector_type(8)));
typedef _Float16 v16h  __attribute__((ext_vector_type(16)));
typedef float    v8f   __attribute__((ext_vector_type(8)));
typedef float    f4    __attribute__((ext_vector_type(4)));

#define TOKENS 8192
#define KDIM   4096
#define ODIM   4096

// ---------------- kernel 1: per-token symmetric int8 quantization (stored f16) ----
// x_q integers in [-127,127] are exactly representable in f16.
__global__ __launch_bounds__(256) void quant_x(const float* __restrict__ x,
                                               _Float16* __restrict__ xq,
                                               float* __restrict__ xs) {
  __shared__ float red[256];
  const int t   = blockIdx.x;
  const int tid = threadIdx.x;
  const float* row = x + (size_t)t * KDIM;

  f4 v[4];
  const f4* rv = (const f4*)row + tid * 4;   // 16 floats per thread
  float m = 0.f;
#pragma unroll
  for (int i = 0; i < 4; ++i) {
    v[i] = rv[i];
#pragma unroll
    for (int j = 0; j < 4; ++j) m = fmaxf(m, fabsf(v[i][j]));
  }
  red[tid] = m;
  __syncthreads();
  for (int s = 128; s > 0; s >>= 1) {
    if (tid < s) red[tid] = fmaxf(red[tid], red[tid + s]);
    __syncthreads();
  }
  const float scale = red[0] * (1.0f / 127.0f);   // reference x_scale (unclamped)
  const float inv   = 1.0f / fmaxf(scale, 1e-8f); // 1 / safe_scale
  if (tid == 0) xs[t] = scale;

  half8 q[2];
#pragma unroll
  for (int i = 0; i < 4; ++i)
#pragma unroll
    for (int j = 0; j < 4; ++j) {
      float qq = fminf(fmaxf(rintf(v[i][j] * inv), -127.f), 127.f);
      q[(i * 4 + j) / 8][(i * 4 + j) % 8] = (_Float16)qq;
    }
  half8* o = (half8*)(xq + (size_t)t * KDIM) + tid * 2;
  o[0] = q[0];
  o[1] = q[1];
}

// ---------------- kernel 2: fold per-K weight_scale into int8 weights -> f16 ------
__global__ __launch_bounds__(256) void deq_w(const int8_t* __restrict__ w8,
                                             const float* __restrict__ wsc,
                                             _Float16* __restrict__ wh) {
  const size_t i = ((size_t)blockIdx.x * 256 + threadIdx.x) * 16;
  const int k = (int)(i & (KDIM - 1));           // K is pow2; 16-elem span stays in-row
  union { int4 q; int8_t b[16]; } u;
  u.q = *(const int4*)(w8 + i);
  f4 s[4];
  const f4* sp = (const f4*)(wsc + k);
#pragma unroll
  for (int j = 0; j < 4; ++j) s[j] = sp[j];
  half8 h[2];
#pragma unroll
  for (int j = 0; j < 16; ++j)
    h[j / 8][j % 8] = (_Float16)((float)u.b[j] * s[j / 4][j % 4]);
  half8* o = (half8*)(wh + i);
  o[0] = h[0];
  o[1] = h[1];
}

// ---------------- kernel 3: f16 WMMA GEMM with fused dequant epilogue -------------
// Block tile 256(M) x 128(N), 8 waves arranged 4(M) x 2(N): each wave owns a
// 64x64 tile = 4x4 accumulators (128 VGPRs). 16 WMMAs per 16 b128 fragment
// loads per K=32 step -> 32.8 FLOP/B of fragment traffic (1.5x better than a
// 64x32 wave tile), keeping the WMMA pipe, not the L0 path, as the limiter.
// Single base pointer per matrix; mi/k displacements become 24-bit instruction
// offsets (mi stride = 16*KDIM*2 = 128 KB < 8 MB).
__global__ __launch_bounds__(256) void gemm_wmma(
    const _Float16* __restrict__ A,   // [TOKENS, KDIM] x_q (f16, integer-valued)
    const _Float16* __restrict__ B,   // [ODIM,   KDIM] w_q * w_scale (f16)
    const float* __restrict__ xs,     // [TOKENS]
    const float* __restrict__ bias,   // [ODIM]
    float* __restrict__ out) {        // [TOKENS, ODIM]
  const int lane  = threadIdx.x & 31;
  const int wave  = threadIdx.x >> 5;
  const int wm    = wave & 3;                 // 4 waves along M (64 rows each)
  const int wn    = wave >> 2;                // 2 waves along N (64 cols each)
  const int m0    = blockIdx.y * 256 + wm * 64;
  const int n0    = blockIdx.x * 128 + wn * 64;
  const int lrow  = lane & 15;
  const int lhalf = lane >> 4;                // K sub-octet select per ISA layout

  const _Float16* aBase = A + (size_t)(m0 + lrow) * KDIM + lhalf * 8;
  const _Float16* bBase = B + (size_t)(n0 + lrow) * KDIM + lhalf * 8;

  v8f zero = {};
  v8f c[4][4];
#pragma unroll
  for (int mi = 0; mi < 4; ++mi)
#pragma unroll
    for (int ni = 0; ni < 4; ++ni) c[mi][ni] = zero;

  for (int k = 0; k < KDIM; k += 32) {
    union { v16h v; half8 h[2]; } a[4], b[4];
#pragma unroll
    for (int mi = 0; mi < 4; ++mi) {
      const _Float16* p = aBase + (size_t)mi * 16 * KDIM + k;
      a[mi].h[0] = *(const half8*)(p);
      a[mi].h[1] = *(const half8*)(p + 16);
      __builtin_prefetch(p + 128, 0, 1);      // global_prefetch next K tiles
    }
#pragma unroll
    for (int ni = 0; ni < 4; ++ni) {
      const _Float16* p = bBase + (size_t)ni * 16 * KDIM + k;
      b[ni].h[0] = *(const half8*)(p);
      b[ni].h[1] = *(const half8*)(p + 16);
      __builtin_prefetch(p + 128, 0, 1);
    }
#pragma unroll
    for (int mi = 0; mi < 4; ++mi)
#pragma unroll
      for (int ni = 0; ni < 4; ++ni)
        c[mi][ni] = __builtin_amdgcn_wmma_f32_16x16x32_f16(
            false, a[mi].v, false, b[ni].v, (short)0, c[mi][ni], false, false);
  }

  // Epilogue: out = acc * x_scale[row] + bias[col]
  // C layout: VGPR r of lane L -> (M = r + 8*(L>=16), N = L&15)
#pragma unroll
  for (int mi = 0; mi < 4; ++mi) {
    float srow[8];
#pragma unroll
    for (int r = 0; r < 8; ++r) srow[r] = xs[m0 + mi * 16 + lhalf * 8 + r];
#pragma unroll
    for (int ni = 0; ni < 4; ++ni) {
      const int col = n0 + ni * 16 + lrow;
      const float bb = bias[col];
#pragma unroll
      for (int r = 0; r < 8; ++r) {
        const int row = m0 + mi * 16 + lhalf * 8 + r;
        out[(size_t)row * ODIM + col] = c[mi][ni][r] * srow[r] + bb;
      }
    }
  }
}

extern "C" void kernel_launch(void* const* d_in, const int* in_sizes, int n_in,
                              void* d_out, int out_size, void* d_ws, size_t ws_size,
                              hipStream_t stream) {
  (void)in_sizes; (void)n_in; (void)out_size; (void)ws_size;
  const float*  x    = (const float*)d_in[0];
  const int8_t* w8   = (const int8_t*)d_in[1];   // int8 buffer per reference dtype
  const float*  wsc  = (const float*)d_in[2];
  const float*  bias = (const float*)d_in[3];
  float* out = (float*)d_out;

  char* ws = (char*)d_ws;
  _Float16* xq = (_Float16*)ws;                                         // 64 MB
  _Float16* wh = (_Float16*)(ws + (size_t)TOKENS * KDIM * 2);           // 32 MB
  float*    xs = (float*)(ws + (size_t)TOKENS * KDIM * 2
                             + (size_t)ODIM * KDIM * 2);                // 32 KB

  quant_x<<<TOKENS, 256, 0, stream>>>(x, xq, xs);
  deq_w<<<(int)(((size_t)ODIM * KDIM) / (256 * 16)), 256, 0, stream>>>(w8, wsc, wh);
  gemm_wmma<<<dim3(ODIM / 128, TOKENS / 256), 256, 0, stream>>>(xq, wh, xs, bias, out);
}
